// WriteHead_52407190946069
// MI455X (gfx1250) — compile-verified, matrix operand order checked by reading, not run on hardware
//
#include <hip/hip_runtime.h>
#include <math.h>

// Problem constants (from reference)
#define BB 128
#define HH 4
#define NN 4096
#define WW 128
#define CTRL (HH * (3 * WW + 6))   // 1560 floats per batch row
#define EPS 1e-8f

typedef __attribute__((ext_vector_type(16))) _Float16 v16h;
typedef __attribute__((ext_vector_type(8)))  float    v8f;

// ---------- small device helpers ----------
__device__ __forceinline__ float sigmoidf_(float x) { return 1.f / (1.f + __expf(-x)); }
__device__ __forceinline__ float softplusf_(float x) {
    return (x > 20.f) ? x : log1pf(expf(x));
}
__device__ __forceinline__ float fastrcp_(float x) { return __builtin_amdgcn_rcpf(x); }

// A/B fragment K-index mapping for V_WMMA_*_16X16X32 (16-bit operands, wave32):
//   lanes 0-15 hold K {0..7, 16..23}; lanes 16-31 hold K {8..15, 24..31}
//   element e<8  -> K = khalf*8 + e
//   element e>=8 -> K = khalf*8 + e + 8
__device__ __forceinline__ int wmma_k_of(int khalf, int e) {
    return khalf * 8 + ((e < 8) ? e : (e + 8));
}

// =====================================================================
// Kernel A: per-(b,h) control preprocessing.
// 512 blocks x 32 threads (one wave each).
//   scal[.+0]=beta [.+1]=gate [.+2]=gamma [.+3]=||key|| [.+4..6]=shift softmax
// =====================================================================
__global__ void ntm_controls_kernel(const float* __restrict__ controls,
                                    float* __restrict__ keys,
                                    float* __restrict__ writev,
                                    float* __restrict__ erasev,
                                    float* __restrict__ scal) {
    const int bh = blockIdx.x;
    const int b  = bh / HH;
    const int h  = bh % HH;
    const int t  = threadIdx.x;
    const float* cb = controls + (size_t)b * CTRL;

    float nacc = 0.f;
#pragma unroll
    for (int j = 0; j < 4; ++j) {
        const int w = t + 32 * j;
        const float k = tanhf(cb[h * WW + w]);
        keys[(size_t)bh * WW + w] = k;
        nacc += k * k;
        erasev[(size_t)bh * WW + w] = sigmoidf_(cb[HH * WW + h * WW + w]);
        writev[(size_t)bh * WW + w] = tanhf(cb[2 * HH * WW + h * WW + w]);
    }
#pragma unroll
    for (int off = 16; off >= 1; off >>= 1) nacc += __shfl_xor(nacc, off);

    if (t == 0) {
        const int base = 3 * HH * WW;
        const float beta  = softplusf_(cb[base + h]);
        const float gate  = sigmoidf_(cb[base + HH + h]);
        float s0 = cb[base + 2 * HH + h * 3 + 0];
        float s1 = cb[base + 2 * HH + h * 3 + 1];
        float s2 = cb[base + 2 * HH + h * 3 + 2];
        const float smax = fmaxf(s0, fmaxf(s1, s2));
        s0 = expf(s0 - smax); s1 = expf(s1 - smax); s2 = expf(s2 - smax);
        const float sinv = 1.f / (s0 + s1 + s2);
        const float gamma = 1.f + softplusf_(cb[base + 2 * HH + 3 * HH + h]);
        float* sc = scal + (size_t)bh * 8;
        sc[0] = beta; sc[1] = gate; sc[2] = gamma; sc[3] = sqrtf(nacc);
        sc[4] = s0 * sinv; sc[5] = s1 * sinv; sc[6] = s2 * sinv;
    }
}

// =====================================================================
// Kernel P: pre-pack WMMA B fragments (keys + write_data) in the exact
// per-lane VGPR layout of V_WMMA_F32_16X16X32_F16. Zero-padding of the
// 4 valid head columns -> 16 via clamped-address loads * 0/1 mask (no
// divergent branches around loads).
// grid = B x 32 threads.
// =====================================================================
__global__ void ntm_pack_kernel(const float* __restrict__ keys,
                                const float* __restrict__ writev,
                                v16h* __restrict__ kpack,
                                v16h* __restrict__ wpack) {
    const int b = blockIdx.x;
    const int l = threadIdx.x;
    const int col   = l & 15;
    const int khalf = l >> 4;

    const float colmask = (col < HH) ? 1.f : 0.f;
    const int   colc    = col & (HH - 1);

    // keys fragments: columns = heads, K = w
#pragma unroll
    for (int kc = 0; kc < 4; ++kc) {
        v16h f;
#pragma unroll
        for (int e = 0; e < 16; ++e) {
            const int w = kc * 32 + wmma_k_of(khalf, e);
            f[e] = (_Float16)(keys[((size_t)(b * HH + colc)) * WW + w] * colmask);
        }
        kpack[((size_t)(b * 4 + kc)) * 32 + l] = f;
    }

    // write_data fragments: columns = w within tile, K = head (pad 4->32)
#pragma unroll
    for (int wt = 0; wt < WW / 16; ++wt) {
        v16h f;
#pragma unroll
        for (int e = 0; e < 16; ++e) {
            const int K     = wmma_k_of(khalf, e);
            const float km  = (K < HH) ? 1.f : 0.f;
            const int   Kc  = K & (HH - 1);
            f[e] = (_Float16)(writev[((size_t)(b * HH + Kc)) * WW + wt * 16 + col] * km);
        }
        wpack[((size_t)(b * 8 + wt)) * 32 + l] = f;
    }
}

// =====================================================================
// Kernel B: scores[b,h,n] = beta * dot(mem[b,n,:],key[b,h,:]) / (|k||m|+eps)
// One wave per (b, 16-row n-tile); memory tile doubles as WMMA A frag and
// row-norm source. Fast v_rcp for the normalization (feeds a softmax).
// grid = (N/16, B) x 32 threads.
// =====================================================================
__global__ void ntm_scores_kernel(const float* __restrict__ mem,
                                  const v16h* __restrict__ kpack,
                                  const float* __restrict__ scal,
                                  float* __restrict__ scores) {
    const int b  = blockIdx.y;
    const int n0 = blockIdx.x * 16;
    const int l  = threadIdx.x;
    const int row   = l & 15;
    const int khalf = l >> 4;

    __shared__ float mn[16];

    const float* mrow = mem + ((size_t)b * NN + n0 + row) * WW;

    v16h a[4];
    float nacc = 0.f;
#pragma unroll
    for (int kc = 0; kc < 4; ++kc) {
        const int base = kc * 32 + khalf * 8;
        const float4 p0 = *(const float4*)(mrow + base);
        const float4 p1 = *(const float4*)(mrow + base + 4);
        const float4 q0 = *(const float4*)(mrow + base + 16);
        const float4 q1 = *(const float4*)(mrow + base + 20);
        a[kc][0] = (_Float16)p0.x; a[kc][1] = (_Float16)p0.y;
        a[kc][2] = (_Float16)p0.z; a[kc][3] = (_Float16)p0.w;
        a[kc][4] = (_Float16)p1.x; a[kc][5] = (_Float16)p1.y;
        a[kc][6] = (_Float16)p1.z; a[kc][7] = (_Float16)p1.w;
        a[kc][8]  = (_Float16)q0.x; a[kc][9]  = (_Float16)q0.y;
        a[kc][10] = (_Float16)q0.z; a[kc][11] = (_Float16)q0.w;
        a[kc][12] = (_Float16)q1.x; a[kc][13] = (_Float16)q1.y;
        a[kc][14] = (_Float16)q1.z; a[kc][15] = (_Float16)q1.w;
        nacc += p0.x*p0.x + p0.y*p0.y + p0.z*p0.z + p0.w*p0.w
              + p1.x*p1.x + p1.y*p1.y + p1.z*p1.z + p1.w*p1.w
              + q0.x*q0.x + q0.y*q0.y + q0.z*q0.z + q0.w*q0.w
              + q1.x*q1.x + q1.y*q1.y + q1.z*q1.z + q1.w*q1.w;
    }
    // full row-norm^2 lives split across lane pair (l, l^16)
    nacc += __shfl_xor(nacc, 16);
    if (l < 16) mn[row] = sqrtf(nacc);
    __syncthreads();

    v8f c = {};
#pragma unroll
    for (int kc = 0; kc < 4; ++kc) {
        const v16h kf = kpack[((size_t)(b * 4 + kc)) * 32 + l];
        c = __builtin_amdgcn_wmma_f32_16x16x32_f16(
                false, a[kc], false, kf, (short)0, c, false, false);
    }

    // C layout: VGPR r -> M=r (lanes 0-15) / M=r+8 (lanes 16-31); N=lane&15
    const int h = l & 15;
    if (h < HH) {
        const float knorm = scal[((size_t)(b * HH + h)) * 8 + 3];
        const float beta  = scal[((size_t)(b * HH + h)) * 8 + 0];
        const int mbase = (l >= 16) ? 8 : 0;
        float s[8];
#pragma unroll
        for (int r = 0; r < 8; ++r)
            s[r] = c[r] * beta * fastrcp_(knorm * mn[mbase + r] + EPS);
        float* sp = scores + ((size_t)(b * HH + h)) * NN + n0 + mbase;
        float4 v0; v0.x = s[0]; v0.y = s[1]; v0.z = s[2]; v0.w = s[3];
        float4 v1; v1.x = s[4]; v1.y = s[5]; v1.z = s[6]; v1.w = s[7];
        *(float4*)(sp)     = v0;
        *(float4*)(sp + 4) = v1;
    }
}

// =====================================================================
// Kernel C: per (b,h): softmax(scores), softmax(bias), gate interp,
// circular shift, gamma-sharpen, renormalize. Hardware transcendentals.
// grid = B*H blocks x 256 threads; 16 elements/thread; LDS ring buffer.
// =====================================================================
__global__ void ntm_dist_kernel(const float* __restrict__ scores,
                                const float* __restrict__ bias,
                                const float* __restrict__ scal,
                                float* __restrict__ wdist) {
    __shared__ float wbuf[NN];
    __shared__ float pbuf[NN];
    __shared__ float red[256];

    const int bh = blockIdx.x;
    const int h  = bh % HH;
    const int t  = threadIdx.x;

    const float* sc = scores + (size_t)bh * NN;
    const float* bi = bias + (size_t)h * NN;
    const float gate  = scal[(size_t)bh * 8 + 1];
    const float gamma = scal[(size_t)bh * 8 + 2];
    const float s0 = scal[(size_t)bh * 8 + 4];
    const float s1 = scal[(size_t)bh * 8 + 5];
    const float s2 = scal[(size_t)bh * 8 + 6];

    float sv[16], bv[16];
    float lmax = -3.4e38f, lbmax = -3.4e38f;
#pragma unroll
    for (int i = 0; i < 16; ++i) {
        const int n = i * 256 + t;
        sv[i] = sc[n]; bv[i] = bi[n];
        lmax = fmaxf(lmax, sv[i]); lbmax = fmaxf(lbmax, bv[i]);
    }
    red[t] = lmax; __syncthreads();
    for (int s = 128; s > 0; s >>= 1) { if (t < s) red[t] = fmaxf(red[t], red[t + s]); __syncthreads(); }
    const float smax = red[0]; __syncthreads();
    red[t] = lbmax; __syncthreads();
    for (int s = 128; s > 0; s >>= 1) { if (t < s) red[t] = fmaxf(red[t], red[t + s]); __syncthreads(); }
    const float bmax = red[0]; __syncthreads();

    float esum = 0.f, bsum = 0.f;
#pragma unroll
    for (int i = 0; i < 16; ++i) {
        const int n = i * 256 + t;
        const float e  = __expf(sv[i] - smax);
        const float pb = __expf(bv[i] - bmax);
        wbuf[n] = e; pbuf[n] = pb;
        esum += e; bsum += pb;
    }
    red[t] = esum; __syncthreads();
    for (int s = 128; s > 0; s >>= 1) { if (t < s) red[t] += red[t + s]; __syncthreads(); }
    const float einv = fastrcp_(red[0]); __syncthreads();
    red[t] = bsum; __syncthreads();
    for (int s = 128; s > 0; s >>= 1) { if (t < s) red[t] += red[t + s]; __syncthreads(); }
    const float binv = fastrcp_(red[0]); __syncthreads();

#pragma unroll
    for (int i = 0; i < 16; ++i) {
        const int n = i * 256 + t;
        wbuf[n] = gate * wbuf[n] * einv + (1.f - gate) * pbuf[n] * binv;
    }
    __syncthreads();

    float sh[16];
    float ssum = 0.f;
#pragma unroll
    for (int i = 0; i < 16; ++i) {
        const int n = i * 256 + t;
        const float wm = wbuf[(n + NN - 1) & (NN - 1)];
        const float wc = wbuf[n];
        const float wp = wbuf[(n + 1) & (NN - 1)];
        const float wsft = s0 * wm + s1 * wc + s2 * wp;
        sh[i] = __powf(wsft, gamma);
        ssum += sh[i];
    }
    red[t] = ssum; __syncthreads();
    for (int s = 128; s > 0; s >>= 1) { if (t < s) red[t] += red[t + s]; __syncthreads(); }
    const float tinv = fastrcp_(red[0] + EPS); __syncthreads();

    float* wd = wdist + (size_t)bh * NN;
#pragma unroll
    for (int i = 0; i < 16; ++i) wd[i * 256 + t] = sh[i] * tinv;
}

// =====================================================================
// Kernel E: out[b,n,w] = mem * prod_h(1 - wd[h,n]*erase[h,w]) + (wd^T@write)
// One wave per (b, 16-row n-tile). WMMA produces the update tile which is
// bounced through LDS so the global stream is row-major float4 per lane:
// each global instr covers one full 512 B row (full-cacheline writes).
// grid = (N/16, B) x 32 threads.
// =====================================================================
__global__ void ntm_update_kernel(const float* __restrict__ mem,
                                  const v16h* __restrict__ wpack,
                                  const float* __restrict__ erasev,
                                  const float* __restrict__ wdist,
                                  float* __restrict__ out) {
    __shared__ float wdl[HH * 16];    // [h][m] write_dist tile
    __shared__ float ebuf[HH * WW];   // [h][w] erase vectors (2 KB)
    __shared__ float upd[16 * WW];    // [m][w] WMMA update tile (8 KB)

    const int b  = blockIdx.y;
    const int n0 = blockIdx.x * 16;
    const int l  = threadIdx.x;
    const int row   = l & 15;
    const int khalf = l >> 4;
    const int col   = row;
    const int wq    = l * 4;          // this lane's 4 contiguous w's

    // stage write_dist tile + erase vectors in LDS
    for (int idx = l; idx < HH * 16; idx += 32) {
        const int hh = idx >> 4, mm = idx & 15;
        wdl[idx] = wdist[((size_t)(b * HH + hh)) * NN + n0 + mm];
    }
#pragma unroll
    for (int hh = 0; hh < HH; ++hh) {
        const float4 ev = *(const float4*)(erasev + ((size_t)(b * HH + hh)) * WW + wq);
        *(float4*)(ebuf + hh * WW + wq) = ev;
    }
    __syncthreads();

    // warm caches for the rows this wave streams
    __builtin_prefetch(mem + ((size_t)b * NN + n0 + row) * WW + khalf * 64, 0, 3);

    // A fragment: rows = n-tile rows, K = head (only khalf==0, e<4 nonzero)
    const float amask = (khalf == 0) ? 1.f : 0.f;
    v16h a;
#pragma unroll
    for (int e = 0; e < 16; ++e) a[e] = (_Float16)0.f;
#pragma unroll
    for (int e = 0; e < HH; ++e)
        a[e] = (_Float16)(wdl[e * 16 + row] * amask);

    const int mbase = (l >= 16) ? 8 : 0;

    // 8 WMMAs -> scatter update tile into LDS
#pragma unroll
    for (int wt = 0; wt < WW / 16; ++wt) {
        const v16h bf = wpack[((size_t)(b * 8 + wt)) * 32 + l];
        v8f c = {};
        c = __builtin_amdgcn_wmma_f32_16x16x32_f16(
                false, a, false, bf, (short)0, c, false, false);
#pragma unroll
        for (int r = 0; r < 8; ++r)
            upd[(mbase + r) * WW + wt * 16 + col] = c[r];
    }
    __syncthreads();

    // loop-invariant erase values for this lane's 4 columns
    float eh[HH][4];
#pragma unroll
    for (int hh = 0; hh < HH; ++hh) {
        const float4 ev = *(const float4*)(ebuf + hh * WW + wq);
        eh[hh][0] = ev.x; eh[hh][1] = ev.y; eh[hh][2] = ev.z; eh[hh][3] = ev.w;
    }

    const float* mrow = mem + ((size_t)b * NN + n0) * WW;
    float*       orow = out + ((size_t)b * NN + n0) * WW;

    // stream 16 rows; each iteration = one full 512 B row across the wave
#pragma unroll 4
    for (int i = 0; i < 16; ++i) {
        float wdr[HH];
#pragma unroll
        for (int hh = 0; hh < HH; ++hh) wdr[hh] = wdl[hh * 16 + i];

        const float4 mv4 = *(const float4*)(mrow + (size_t)i * WW + wq);
        const float4 uv4 = *(const float4*)(upd + i * WW + wq);
        const float mv[4] = {mv4.x, mv4.y, mv4.z, mv4.w};
        const float uv[4] = {uv4.x, uv4.y, uv4.z, uv4.w};
        float ov[4];
#pragma unroll
        for (int j = 0; j < 4; ++j) {
            float prod = 1.f;
#pragma unroll
            for (int hh = 0; hh < HH; ++hh)
                prod = prod * (1.f - wdr[hh] * eh[hh][j]);
            ov[j] = mv[j] * prod + uv[j];
        }
        float4 ov4; ov4.x = ov[0]; ov4.y = ov[1]; ov4.z = ov[2]; ov4.w = ov[3];
        *(float4*)(orow + (size_t)i * WW + wq) = ov4;
    }
}

// =====================================================================
// Host launch
// =====================================================================
extern "C" void kernel_launch(void* const* d_in, const int* in_sizes, int n_in,
                              void* d_out, int out_size, void* d_ws, size_t ws_size,
                              hipStream_t stream) {
    (void)in_sizes; (void)n_in; (void)out_size; (void)ws_size;

    const float* memory   = (const float*)d_in[0];   // [B,N,W]
    const float* controls = (const float*)d_in[1];   // [B, H*(3W+6)]
    const float* bias     = (const float*)d_in[2];   // [1,H,N]
    float*       out      = (float*)d_out;           // [B,N,W]

    float* wsf = (float*)d_ws;
    float* keys   = wsf;                                      // B*H*W
    float* writev = keys   + (size_t)BB * HH * WW;            // B*H*W
    float* erasev = writev + (size_t)BB * HH * WW;            // B*H*W
    float* scal   = erasev + (size_t)BB * HH * WW;            // B*H*8
    float* scores = scal   + (size_t)BB * HH * 8;             // B*H*N
    float* wdist  = scores + (size_t)BB * HH * NN;            // B*H*N
    float* packs  = wdist  + (size_t)BB * HH * NN;
    v16h*  kpack  = (v16h*)packs;                             // B*4*32 v16h (512 KB)
    v16h*  wpack  = (v16h*)(packs + (size_t)BB * 4 * 32 * 8); // B*8*32 v16h (1 MB)

    ntm_controls_kernel<<<dim3(BB * HH), dim3(32), 0, stream>>>(
        controls, keys, writev, erasev, scal);

    ntm_pack_kernel<<<dim3(BB), dim3(32), 0, stream>>>(
        keys, writev, kpack, wpack);

    ntm_scores_kernel<<<dim3(NN / 16, BB), dim3(32), 0, stream>>>(
        memory, kpack, scal, scores);

    ntm_dist_kernel<<<dim3(BB * HH), dim3(256), 0, stream>>>(
        scores, bias, scal, wdist);

    ntm_update_kernel<<<dim3(NN / 16, BB), dim3(32), 0, stream>>>(
        memory, wpack, erasev, wdist, out);
}
